// MambaLMHeadModel_46746424050470
// MI455X (gfx1250) — matrix operand order repaired and checked by reading, not compiled, not run on hardware
//
#include <hip/hip_runtime.h>
#include <hip/hip_bf16.h>

// ---------------- model dims ----------------
#define T_SEQ    1024
#define D_MODEL  1024
#define N_LAYER  2
#define VOCAB    50257
#define D_STATE  64
#define D_CONV   4
#define D_INNER  2048
#define NHEADS   32
#define HEADDIM  64
#define CONV_DIM 2176            // D_INNER + 2*D_STATE
#define D_PROJ   4256            // 2*D_INNER + 2*D_STATE + NHEADS
#define D_FF     4096
#define EPSV     1e-5f

// ---------------- vector types ----------------
typedef __attribute__((ext_vector_type(16))) __bf16 v16bf;
typedef __attribute__((ext_vector_type(8)))  __bf16 v8bf;
typedef __attribute__((ext_vector_type(8)))  float  v8f;
typedef __attribute__((ext_vector_type(4)))  float  v4f;

__device__ __forceinline__ float silu_f(float x) {
  return x / (1.f + __expf(-x));
}

// ---------------- CDNA5 async-to-LDS helpers (inline asm, ISA ch.10/15.18) ----
// per-lane: LDS[lds_off] <- 16B from global ptr; tracked by ASYNCcnt
__device__ __forceinline__ void async_copy_b128(unsigned lds_off, const void* g) {
  asm volatile("global_load_async_to_lds_b128 %0, %1, off"
               :: "v"(lds_off), "v"(g) : "memory");
}
__device__ __forceinline__ void wait_async_0() {
  asm volatile("s_wait_asynccnt 0x0" ::: "memory");
}

// LDS byte offset of a shared pointer = low 32 bits of the flat address
__device__ __forceinline__ unsigned lds_off_of(const void* p) {
  return (unsigned)(uintptr_t)p;
}

// 16x f32 (two 32B runs in LDS) -> one bf16 fragment (v_cvt packs)
__device__ __forceinline__ v16bf frag_cvt(const float* p0, const float* p1) {
  v4f a = *(const v4f*)p0, b = *(const v4f*)(p0 + 4);
  v4f c = *(const v4f*)p1, d = *(const v4f*)(p1 + 4);
  v16bf r;
  r[0]  = (__bf16)a.x; r[1]  = (__bf16)a.y; r[2]  = (__bf16)a.z; r[3]  = (__bf16)a.w;
  r[4]  = (__bf16)b.x; r[5]  = (__bf16)b.y; r[6]  = (__bf16)b.z; r[7]  = (__bf16)b.w;
  r[8]  = (__bf16)c.x; r[9]  = (__bf16)c.y; r[10] = (__bf16)c.z; r[11] = (__bf16)c.w;
  r[12] = (__bf16)d.x; r[13] = (__bf16)d.y; r[14] = (__bf16)d.z; r[15] = (__bf16)d.w;
  return r;
}

// =======================================================================
// WMMA GEMM:  C[M,N] = A[M,K] @ W[N,K]^T
// f32 tiles stream HBM -> LDS via global_load_async_to_lds_b128 (ASYNCcnt),
// double-buffered; fragments are gather-converted to bf16 at use and fed to
// v_wmma_f32_16x16x32_bf16 (co-executes with the cvt VALU work).
// 256 threads = 8 waves (4 along M x 2 along N); block tile 128x64, BK=32;
// wave tile 32x32 -> 4 WMMAs / wave / K-iter.
// =======================================================================
#define BM 128
#define BN 64
#define BK 32
#define ASTR 36   // f32 per LDS row (144B: 16B-aligned, bank-skewed)

__global__ __launch_bounds__(256)
void gemm_wmma_kernel(const float* __restrict__ A, const float* __restrict__ W,
                      float* __restrict__ C, int M, int N, int K) {
  __shared__ __align__(16) float As[2][BM * ASTR];   // 2 x 18432 B
  __shared__ __align__(16) float Ws[2][BN * ASTR];   // 2 x  9216 B

  const int tid   = threadIdx.x;
  const int lane  = tid & 31;
  const int wv    = tid >> 5;
  const int wm    = wv & 3;          // wave row: 32 rows each
  const int wn    = wv >> 2;         // wave col: 32 cols each
  const int bm    = blockIdx.y * BM;
  const int bn    = blockIdx.x * BN;
  const int halfw = lane >> 4;       // 0/1
  const int lrow  = lane & 15;
  const int kb    = halfw * 8;       // A-fragment K base within 32-chunk

  v8f acc[2][2] = {{{}, {}}, {{}, {}}};

  // issue one K-tile's async copies into buffer p
  auto issue_tile = [&](int k0, int p) {
    // A tile: 128 rows x 8 chunks of 16B = 1024 lane-tasks (4 per thread)
#pragma unroll
    for (int r = 0; r < 4; ++r) {
      const int task = tid + 256 * r;
      const int row  = task >> 3;
      const int ch   = task & 7;
      async_copy_b128(lds_off_of(&As[p][row * ASTR + ch * 4]),
                      A + (size_t)(bm + row) * K + k0 + ch * 4);
    }
    // W tile: 64 rows x 8 chunks (2 per thread); zero-fill N tail
#pragma unroll
    for (int r = 0; r < 2; ++r) {
      const int task = tid + 256 * r;
      const int row  = task >> 3;
      const int ch   = task & 7;
      const int gn   = bn + row;
      if (gn < N) {
        async_copy_b128(lds_off_of(&Ws[p][row * ASTR + ch * 4]),
                        W + (size_t)gn * K + k0 + ch * 4);
      } else {
        *reinterpret_cast<v4f*>(&Ws[p][row * ASTR + ch * 4]) = (v4f){0.f, 0.f, 0.f, 0.f};
      }
    }
  };

  issue_tile(0, 0);
  wait_async_0();
  __syncthreads();

  const int nIter = K / BK;
  for (int it = 0; it < nIter; ++it) {
    const int p = it & 1;
    if (it + 1 < nIter) issue_tile((it + 1) * BK, p ^ 1);

    const float* Ab = As[p];
    const float* Wb = Ws[p];

    // A fragments: lane row = wm*32+mf*16+lrow; K elems {kb..kb+7, 16+kb..}
    v16bf af[2];
#pragma unroll
    for (int mf = 0; mf < 2; ++mf) {
      const int row = wm * 32 + mf * 16 + lrow;
      af[mf] = frag_cvt(&Ab[row * ASTR + kb], &Ab[row * ASTR + 16 + kb]);
    }
    // B fragments: lane col = wn*32+nf*16+lrow; K elems {kb..kb+7, 16+kb..}
    v16bf bfr[2];
#pragma unroll
    for (int nf = 0; nf < 2; ++nf) {
      const int col = wn * 32 + nf * 16 + lrow;
      bfr[nf] = frag_cvt(&Wb[col * ASTR + kb],
                         &Wb[col * ASTR + 16 + kb]);
    }
#pragma unroll
    for (int mf = 0; mf < 2; ++mf)
#pragma unroll
      for (int nf = 0; nf < 2; ++nf)
        acc[mf][nf] = __builtin_amdgcn_wmma_f32_16x16x32_bf16(
            false, af[mf], false, bfr[nf], (short)0, acc[mf][nf], false, false);

    wait_async_0();     // next tile landed in LDS
    __syncthreads();    // all waves done reading buffer p
  }

  // ---- store: C/D layout m = 8*(lane>>4)+r, n = lane&15 ----
#pragma unroll
  for (int mf = 0; mf < 2; ++mf) {
    const int mrow = bm + wm * 32 + mf * 16 + halfw * 8;
#pragma unroll
    for (int nf = 0; nf < 2; ++nf) {
      const int nc = bn + wn * 32 + nf * 16 + lrow;
      if (nc < N) {
#pragma unroll
        for (int r = 0; r < 8; ++r)
          C[(size_t)(mrow + r) * N + nc] = acc[mf][nf][r];
      }
    }
  }
}

// =======================================================================
// Elementwise / norm / scan kernels
// =======================================================================
__global__ void embed_kernel(const int* __restrict__ idx,
                             const float* __restrict__ E,
                             float* __restrict__ h) {
  const int t = blockIdx.x;
  const int v = idx[t];
  for (int d = threadIdx.x; d < D_MODEL; d += 256)
    h[(size_t)t * D_MODEL + d] = E[(size_t)v * D_MODEL + d];
}

__global__ void layernorm_kernel(const float* __restrict__ x,
                                 const float* __restrict__ w,
                                 float* __restrict__ y, int D) {
  const int t = blockIdx.x;
  const float* xr = x + (size_t)t * D;
  float* yr = y + (size_t)t * D;
  __shared__ float red[256];
  float s = 0.f;
  for (int d = threadIdx.x; d < D; d += 256) s += xr[d];
  red[threadIdx.x] = s; __syncthreads();
  for (int o = 128; o > 0; o >>= 1) {
    if (threadIdx.x < o) red[threadIdx.x] += red[threadIdx.x + o];
    __syncthreads();
  }
  const float mu = red[0] / (float)D;
  __syncthreads();
  float v = 0.f;
  for (int d = threadIdx.x; d < D; d += 256) { float z = xr[d] - mu; v += z * z; }
  red[threadIdx.x] = v; __syncthreads();
  for (int o = 128; o > 0; o >>= 1) {
    if (threadIdx.x < o) red[threadIdx.x] += red[threadIdx.x + o];
    __syncthreads();
  }
  const float rs = rsqrtf(red[0] / (float)D + EPSV);
  for (int d = threadIdx.x; d < D; d += 256) yr[d] = (xr[d] - mu) * rs * w[d];
}

__global__ void conv_silu_kernel(const float* __restrict__ proj,
                                 const float* __restrict__ cw,
                                 const float* __restrict__ cb,
                                 float* __restrict__ out) {
  const int t = blockIdx.x;
  for (int c = threadIdx.x; c < CONV_DIM; c += 256) {
    float acc = cb[c];
#pragma unroll
    for (int k = 0; k < D_CONV; ++k) {
      const int ts = t - (D_CONV - 1) + k;
      if (ts >= 0)
        acc += proj[(size_t)ts * D_PROJ + D_INNER + c] * cw[c * D_CONV + k];
    }
    out[(size_t)t * CONV_DIM + c] = silu_f(acc);
  }
}

__global__ void dt_kernel(const float* __restrict__ proj,
                          const float* __restrict__ dt_bias,
                          const float* __restrict__ A_log,
                          float* __restrict__ dtb, float* __restrict__ dAb) {
  const int i = blockIdx.x * 256 + threadIdx.x;  // T*NHEADS
  if (i >= T_SEQ * NHEADS) return;
  const int h = i & (NHEADS - 1);
  const int t = i >> 5;
  const float raw = proj[(size_t)t * D_PROJ + (D_PROJ - NHEADS) + h] + dt_bias[h];
  const float sp = (raw > 20.f) ? raw : log1pf(__expf(raw));  // softplus
  dtb[i] = sp;
  dAb[i] = __expf(-__expf(A_log[h]) * sp);
}

// one persistent block per head; 256 threads = 64 p-rows x 4 n-groups of 16
__global__ __launch_bounds__(256)
void scan_kernel(const float* __restrict__ conv, const float* __restrict__ dtb,
                 const float* __restrict__ dAb, const float* __restrict__ Dp,
                 float* __restrict__ y) {
  const int head = blockIdx.x;
  const int tid  = threadIdx.x;
  const int p    = tid >> 2;
  const int ng   = tid & 3;
  const int lane = tid & 31;

  float s[16];
#pragma unroll
  for (int j = 0; j < 16; ++j) s[j] = 0.f;

  __shared__ float xs[HEADDIM], Bs[D_STATE], Cs[D_STATE], sc[2];
  const float Dh = Dp[head];

  for (int t = 0; t < T_SEQ; ++t) {
    const float* row = conv + (size_t)t * CONV_DIM;
    if (tid < 64)        xs[tid]        = row[head * HEADDIM + tid];
    else if (tid < 128)  Bs[tid - 64]   = row[D_INNER + (tid - 64)];
    else if (tid < 192)  Cs[tid - 128]  = row[D_INNER + D_STATE + (tid - 128)];
    else if (tid == 192) { sc[0] = dtb[t * NHEADS + head]; sc[1] = dAb[t * NHEADS + head]; }
    __syncthreads();

    const float dt_t = sc[0], dA_t = sc[1];
    const float xp   = xs[p];
    const float coef = dt_t * xp;
    float acc = 0.f;
#pragma unroll
    for (int j = 0; j < 16; ++j) {
      const int n = ng * 16 + j;
      s[j] = dA_t * s[j] + coef * Bs[n];
      acc += s[j] * Cs[n];
    }
    acc += __shfl_xor(acc, 1, 32);
    acc += __shfl_xor(acc, 2, 32);
    if ((lane & 3) == 0)
      y[(size_t)t * D_INNER + head * HEADDIM + p] = acc + Dh * xp;
    __syncthreads();
  }
}

// y <- rmsnorm(y * silu(z)) * gw   (in-place safe; one block per row)
__global__ void gate_rmsnorm_kernel(const float* __restrict__ y,
                                    const float* __restrict__ proj,
                                    const float* __restrict__ gw,
                                    float* __restrict__ out) {
  const int t = blockIdx.x;
  __shared__ float u[D_INNER];
  __shared__ float red[256];
  const float* yr = y + (size_t)t * D_INNER;
  const float* zr = proj + (size_t)t * D_PROJ;   // z = first D_INNER cols
  float s = 0.f;
  for (int d = threadIdx.x; d < D_INNER; d += 256) {
    const float uz = yr[d] * silu_f(zr[d]);
    u[d] = uz; s += uz * uz;
  }
  red[threadIdx.x] = s; __syncthreads();
  for (int o = 128; o > 0; o >>= 1) {
    if (threadIdx.x < o) red[threadIdx.x] += red[threadIdx.x + o];
    __syncthreads();
  }
  const float rs = rsqrtf(red[0] / (float)D_INNER + EPSV);
  float* orow = out + (size_t)t * D_INNER;
  for (int d = threadIdx.x; d < D_INNER; d += 256) orow[d] = u[d] * rs * gw[d];
}

__global__ void swiglu_kernel(const float* __restrict__ big, float* __restrict__ out) {
  const size_t i = (size_t)blockIdx.x * 256 + threadIdx.x;  // T*D_FF
  const size_t t = i >> 12;          // /4096
  const size_t d = i & (D_FF - 1);
  const float yv = big[t * (2 * D_FF) + d];
  const float g  = big[t * (2 * D_FF) + D_FF + d];
  out[i] = yv * silu_f(g);
}

__global__ void add2_kernel(float* __restrict__ out, const float* __restrict__ a,
                            const float* __restrict__ b, int n) {
  const int i = blockIdx.x * 256 + threadIdx.x;
  if (i < n) out[i] = a[i] + b[i];
}

__global__ void nll_row_kernel(const float* __restrict__ logits,
                               const int* __restrict__ targets,
                               float* __restrict__ nll) {
  const int t = blockIdx.x;
  const float* row = logits + (size_t)t * VOCAB;
  __shared__ float red[256];
  float m = -1e30f;
  for (int d = threadIdx.x; d < VOCAB; d += 256) m = fmaxf(m, row[d]);
  red[threadIdx.x] = m; __syncthreads();
  for (int o = 128; o > 0; o >>= 1) {
    if (threadIdx.x < o) red[threadIdx.x] = fmaxf(red[threadIdx.x], red[threadIdx.x + o]);
    __syncthreads();
  }
  m = red[0]; __syncthreads();
  float s = 0.f;
  for (int d = threadIdx.x; d < VOCAB; d += 256) s += __expf(row[d] - m);
  red[threadIdx.x] = s; __syncthreads();
  for (int o = 128; o > 0; o >>= 1) {
    if (threadIdx.x < o) red[threadIdx.x] += red[threadIdx.x + o];
    __syncthreads();
  }
  if (threadIdx.x == 0) {
    const float lse = m + __logf(red[0]);
    nll[t] = lse - row[targets[t]];
  }
}

__global__ void loss_reduce_kernel(const float* __restrict__ nll, float* __restrict__ loss) {
  __shared__ float red[256];
  float s = 0.f;
  for (int i = threadIdx.x; i < T_SEQ; i += 256) s += nll[i];
  red[threadIdx.x] = s; __syncthreads();
  for (int o = 128; o > 0; o >>= 1) {
    if (threadIdx.x < o) red[threadIdx.x] += red[threadIdx.x + o];
    __syncthreads();
  }
  if (threadIdx.x == 0) *loss = red[0] / (float)T_SEQ;
}

// =======================================================================
// launcher
// =======================================================================
extern "C" void kernel_launch(void* const* d_in, const int* in_sizes, int n_in,
                              void* d_out, int out_size, void* d_ws, size_t ws_size,
                              hipStream_t stream) {
  const int*   idx        = (const int*)d_in[0];
  const int*   targets    = (const int*)d_in[1];
  const float* E          = (const float*)d_in[2];
  const float* norm_w     = (const float*)d_in[3];
  const float* norm2_w    = (const float*)d_in[4];
  const float* in_proj_w  = (const float*)d_in[5];
  const float* conv_w     = (const float*)d_in[6];
  const float* conv_b     = (const float*)d_in[7];
  const float* dt_bias    = (const float*)d_in[8];
  const float* A_log      = (const float*)d_in[9];
  const float* Dp         = (const float*)d_in[10];
  const float* gnorm_w    = (const float*)d_in[11];
  const float* out_proj_w = (const float*)d_in[12];
  const float* fc1_w      = (const float*)d_in[13];
  const float* fc2_w      = (const float*)d_in[14];
  const float* norm_f_w   = (const float*)d_in[15];

  float* logits = (float*)d_out;
  float* loss   = logits + (size_t)T_SEQ * VOCAB;

  float* ws = (float*)d_ws;
  size_t o = 0;
  float* f_h    = ws + o; o += (size_t)T_SEQ * D_MODEL;
  float* f_xn   = ws + o; o += (size_t)T_SEQ * D_MODEL;
  float* f_res  = ws + o; o += (size_t)T_SEQ * D_MODEL;
  float* f_big  = ws + o; o += (size_t)T_SEQ * (2 * D_FF);   // proj (4256) & mlp1 (8192)
  float* f_conv = ws + o; o += (size_t)T_SEQ * CONV_DIM;
  float* f_y    = ws + o; o += (size_t)T_SEQ * D_INNER;
  float* f_swi  = ws + o; o += (size_t)T_SEQ * D_FF;
  float* f_dt   = ws + o; o += (size_t)T_SEQ * NHEADS;
  float* f_dA   = ws + o; o += (size_t)T_SEQ * NHEADS;
  float* f_nll  = ws + o; o += (size_t)T_SEQ;

  const dim3 blk(256);

  embed_kernel<<<T_SEQ, blk, 0, stream>>>(idx, E, f_h);

  for (int i = 0; i < N_LAYER; ++i) {
    // ---- Mamba2 mixer ----
    layernorm_kernel<<<T_SEQ, blk, 0, stream>>>(f_h, norm_w + i * D_MODEL, f_xn, D_MODEL);
    gemm_wmma_kernel<<<dim3((D_PROJ + BN - 1) / BN, T_SEQ / BM), blk, 0, stream>>>(
        f_xn, in_proj_w + (size_t)i * D_PROJ * D_MODEL, f_big, T_SEQ, D_PROJ, D_MODEL);
    conv_silu_kernel<<<T_SEQ, blk, 0, stream>>>(
        f_big, conv_w + (size_t)i * CONV_DIM * D_CONV, conv_b + (size_t)i * CONV_DIM, f_conv);
    dt_kernel<<<(T_SEQ * NHEADS + 255) / 256, blk, 0, stream>>>(
        f_big, dt_bias + i * NHEADS, A_log + i * NHEADS, f_dt, f_dA);
    scan_kernel<<<NHEADS, blk, 0, stream>>>(f_conv, f_dt, f_dA, Dp + i * NHEADS, f_y);
    gate_rmsnorm_kernel<<<T_SEQ, blk, 0, stream>>>(f_y, f_big, gnorm_w + (size_t)i * D_INNER, f_y);
    gemm_wmma_kernel<<<dim3(D_MODEL / BN, T_SEQ / BM), blk, 0, stream>>>(
        f_y, out_proj_w + (size_t)i * D_MODEL * D_INNER, f_xn, T_SEQ, D_MODEL, D_INNER);
    add2_kernel<<<(T_SEQ * D_MODEL + 255) / 256, blk, 0, stream>>>(
        f_res, f_xn, f_h, T_SEQ * D_MODEL);

    // ---- gated MLP ----
    layernorm_kernel<<<T_SEQ, blk, 0, stream>>>(f_res, norm2_w + i * D_MODEL, f_xn, D_MODEL);
    gemm_wmma_kernel<<<dim3((2 * D_FF) / BN, T_SEQ / BM), blk, 0, stream>>>(
        f_xn, fc1_w + (size_t)i * (2 * D_FF) * D_MODEL, f_big, T_SEQ, 2 * D_FF, D_MODEL);
    swiglu_kernel<<<(T_SEQ * D_FF + 255) / 256, blk, 0, stream>>>(f_big, f_swi);
    gemm_wmma_kernel<<<dim3(D_MODEL / BN, T_SEQ / BM), blk, 0, stream>>>(
        f_swi, fc2_w + (size_t)i * D_MODEL * D_FF, f_xn, T_SEQ, D_MODEL, D_FF);
    add2_kernel<<<(T_SEQ * D_MODEL + 255) / 256, blk, 0, stream>>>(
        f_h, f_xn, f_res, T_SEQ * D_MODEL);
  }

  // ---- head ----
  layernorm_kernel<<<T_SEQ, blk, 0, stream>>>(f_h, norm_f_w, f_xn, D_MODEL);
  gemm_wmma_kernel<<<dim3((VOCAB + BN - 1) / BN, T_SEQ / BM), blk, 0, stream>>>(
      f_xn, E, logits, T_SEQ, VOCAB, D_MODEL);
  nll_row_kernel<<<T_SEQ, blk, 0, stream>>>(logits, targets, f_nll);
  loss_reduce_kernel<<<1, blk, 0, stream>>>(f_nll, loss);
}
